// FP8LinearSlow_5832565588648
// MI455X (gfx1250) — matrix operand order repaired and checked by reading, compiled, not run on hardware
//
#include <hip/hip_runtime.h>
#include <hip/hip_bf16.h>
#include <stdint.h>

typedef __bf16 bf16_t;
typedef __attribute__((ext_vector_type(16))) __bf16 v16bf;
typedef __attribute__((ext_vector_type(8)))  __bf16 v8bf;
typedef __attribute__((ext_vector_type(8)))  float  v8f;
typedef int v4i __attribute__((vector_size(16)));

constexpr int KDIM = 4096;
constexpr int NDIM = 16384;

constexpr int BM = 256;    // tokens per block
constexpr int BN = 128;    // out-features per block
constexpr int BK = 64;     // k-chunk
constexpr int NT = KDIM / BK;
constexpr int LDSP = 72;   // padded LDS row stride (elems); 144 B rows, 16 B aligned
constexpr int TILE_A = BM * LDSP;
constexpr int TILE_B = BN * LDSP;

#if __has_builtin(__builtin_amdgcn_global_load_async_to_lds_b128)
#define HAVE_ASYNC_LDS 1
#endif

typedef __attribute__((address_space(1))) v4i* gv4p;  // global int4*
typedef __attribute__((address_space(3))) v4i* lv4p;  // LDS int4*

// Copy 64 B per lane, global -> LDS. Async DMA (ASYNCcnt).
__device__ __forceinline__ void cp_64B(const bf16_t* g, bf16_t* l) {
#ifdef HAVE_ASYNC_LDS
  gv4p gp = (gv4p)g;
  lv4p lp = (lv4p)l;
  __builtin_amdgcn_global_load_async_to_lds_b128(gp, lp, 0, 0);
  __builtin_amdgcn_global_load_async_to_lds_b128(gp, lp, 16, 0);
  __builtin_amdgcn_global_load_async_to_lds_b128(gp, lp, 32, 0);
  __builtin_amdgcn_global_load_async_to_lds_b128(gp, lp, 48, 0);
#else
  const uint4* s = (const uint4*)g;
  uint4* d = (uint4*)l;
  uint4 t0 = s[0], t1 = s[1], t2 = s[2], t3 = s[3];
  d[0] = t0; d[1] = t1; d[2] = t2; d[3] = t3;
#endif
}

__device__ __forceinline__ void wait_async() {
#ifdef HAVE_ASYNC_LDS
#if __has_builtin(__builtin_amdgcn_s_wait_asynccnt)
  __builtin_amdgcn_s_wait_asynccnt(0);
#else
  asm volatile("s_wait_asynccnt 0x0" ::: "memory");
#endif
#endif
}

// Dequantize 8 fp8-e4m3fn bytes to v8bf. (e,m) goes straight into the f32
// exponent/mantissa field; 2^120 is folded into the scale -> exact for
// normals AND e4m3 denormals, one mul per element (f32, matching reference).
__device__ __forceinline__ v8bf dq8(uint32_t q0, uint32_t q1, float s120) {
  v8bf r;
#pragma unroll
  for (int b = 0; b < 4; ++b) {
    uint32_t v0 = (q0 >> (8 * b)) & 0xffu;
    uint32_t h0 = ((v0 & 0x80u) << 24) | ((v0 & 0x7fu) << 20);
    r[b] = (__bf16)(__uint_as_float(h0) * s120);
    uint32_t v1 = (q1 >> (8 * b)) & 0xffu;
    uint32_t h1 = ((v1 & 0x80u) << 24) | ((v1 & 0x7fu) << 20);
    r[4 + b] = (__bf16)(__uint_as_float(h1) * s120);
  }
  return r;
}

__global__ __launch_bounds__(512)
void fp8_linear_bf16_wmma(const bf16_t* __restrict__ X,
                          const uint8_t* __restrict__ W,
                          const float*   __restrict__ Sc,
                          const bf16_t*  __restrict__ Bias,
                          bf16_t*        __restrict__ Out) {
  __shared__ bf16_t As[2 * TILE_A];  // x tiles, double buffered (72 KB)
  __shared__ bf16_t Bs[2 * TILE_B];  // dequantized W tiles (36 KB)

  const int tid  = threadIdx.x;      // 0..511 (16 wave32)
  const int lane = tid & 31;
  const int l16  = lane & 15;
  const int lh   = lane >> 4;
  const int wave = tid >> 5;
  const int wm   = wave >> 2;        // 0..3 : 64-row strip
  const int wn   = wave & 3;         // 0..3 : 32-col strip

  const int mBlock = blockIdx.y * BM;
  const int nBlock = blockIdx.x * BN;

  // X loader: 2 threads per row (256 rows), 32 elems (64 B) each
  const int xrow  = tid >> 1;
  const int xhalf = tid & 1;
  // W loader: 4 threads per row (128 rows), 16 fp8 (16 B) each
  const int wrow = tid >> 2;
  const int wseg = tid & 3;

  const float wscale = Sc[nBlock + wrow] * 0x1.0p+120f;

  const bf16_t*  xg = X + (size_t)(mBlock + xrow) * KDIM + xhalf * 32;
  const uint8_t* wg = W + (size_t)(nBlock + wrow) * KDIM + wseg * 16;

  bf16_t* asw = &As[xrow * LDSP + xhalf * 32];
  bf16_t* bsw = &Bs[wrow * LDSP + wseg * 16];

  v8f acc[4][2] = {};

  // ---- pipeline prologue: kick X tile 0, load W tile 0 into regs --------
  cp_64B(xg, asw);
  uint4 wq = *(const uint4*)wg;

  for (int kt = 0; kt < NT; ++kt) {
    const int p = kt & 1;

    // decode current W regs into Bs[p]
    v8bf* bd = (v8bf*)(bsw + p * TILE_B);
    bd[0] = dq8(wq.x, wq.y, wscale);
    bd[1] = dq8(wq.z, wq.w, wscale);

    // issue next W global loads (consumed next iteration)
    uint4 nq = {};
    if (kt + 1 < NT) {
      const uint8_t* wn_ = wg + (kt + 1) * BK;
      __builtin_prefetch(wn_, 0, 3);
      nq = *(const uint4*)wn_;
    }

    wait_async();        // X tile p resident in LDS (per-wave; barrier joins)
    __syncthreads();

    // kick X tile kt+1 into the other buffer (reads of it done by barrier)
    if (kt + 1 < NT) cp_64B(xg + (kt + 1) * BK, asw + ((p ^ 1) * TILE_A));

    const bf16_t* Ab = &As[p * TILE_A];
    const bf16_t* Bb = &Bs[p * TILE_B];

#pragma unroll
    for (int ks = 0; ks < BK; ks += 32) {
      v16bf afrag[4];
#pragma unroll
      for (int i = 0; i < 4; ++i) {
        // A 16x32 layout: row = l16, 16 B chunks at K = 8*lh and 16+8*lh
        const bf16_t* pa = &Ab[(wm * 64 + i * 16 + l16) * LDSP + ks + lh * 8];
        v8bf lo = *(const v8bf*)pa;
        v8bf hi = *(const v8bf*)(pa + 16);
        afrag[i] = __builtin_shufflevector(lo, hi, 0, 1, 2, 3, 4, 5, 6, 7, 8,
                                           9, 10, 11, 12, 13, 14, 15);
      }
      v16bf bfrag[2];
#pragma unroll
      for (int j = 0; j < 2; ++j) {
        // B 32x16 layout: col = l16, 16 B chunks at K = 8*lh and 16+8*lh
        const bf16_t* pb = &Bb[(wn * 32 + j * 16 + l16) * LDSP + ks + lh * 8];
        v8bf lo = *(const v8bf*)pb;
        v8bf hi = *(const v8bf*)(pb + 16);
        bfrag[j] = __builtin_shufflevector(lo, hi, 0, 1, 2, 3, 4, 5, 6, 7, 8,
                                           9, 10, 11, 12, 13, 14, 15);
      }
#pragma unroll
      for (int i = 0; i < 4; ++i)
#pragma unroll
        for (int j = 0; j < 2; ++j)
          acc[i][j] = __builtin_amdgcn_wmma_f32_16x16x32_bf16(
              false, afrag[i], false, bfrag[j], (short)0, acc[i][j], false,
              false);
    }

    wq = nq;
  }

  // ---- epilogue: + bias, f32 -> bf16, per C/D VGPR layout ---------------
#pragma unroll
  for (int j = 0; j < 2; ++j) {
    const int n   = nBlock + wn * 32 + j * 16 + l16;
    const float b = (float)Bias[n];
#pragma unroll
    for (int i = 0; i < 4; ++i) {
      const int mbase = mBlock + wm * 64 + i * 16 + lh * 8;
      bf16_t* orow = Out + (size_t)mbase * NDIM + n;
#pragma unroll
      for (int r = 0; r < 8; ++r)
        orow[(size_t)r * NDIM] = (bf16_t)(acc[i][j][r] + b);
    }
  }
}

extern "C" void kernel_launch(void* const* d_in, const int* in_sizes, int n_in,
                              void* d_out, int out_size, void* d_ws,
                              size_t ws_size, hipStream_t stream) {
  const bf16_t*  X  = (const bf16_t*)d_in[0];
  const uint8_t* W  = (const uint8_t*)d_in[1];
  const float*   Sc = (const float*)d_in[2];
  const bf16_t*  Bi = (const bf16_t*)d_in[3];
  bf16_t*        O  = (bf16_t*)d_out;

  const int tokens = in_sizes[0] / KDIM;  // 8192
  dim3 grid(NDIM / BN, tokens / BM);      // (128, 32)
  fp8_linear_bf16_wmma<<<grid, 512, 0, stream>>>(X, W, Sc, Bi, O);
}
